// PointerDecoderSort_57578331570396
// MI455X (gfx1250) — matrix operand order repaired and checked by reading, not compiled, not run on hardware
//
#include <hip/hip_runtime.h>
#include <hip/hip_bf16.h>

// ---------------------------------------------------------------------------
// PointerDecoderSort for MI455X (gfx1250, wave32, WMMA)
// B=4096, N=64, D=256, H=8, HD=32, DFF=1024
// One workgroup (8 waves) per batch element; all GEMMs via
// v_wmma_f32_16x16x32_f16. Weights pre-swizzled to WMMA-B operand order so
// every operand fetch is b128-wide (no per-half packing).
// ---------------------------------------------------------------------------

typedef __attribute__((ext_vector_type(16))) _Float16 v16h;
typedef __attribute__((ext_vector_type(8)))  _Float16 v8h;
typedef __attribute__((ext_vector_type(8)))  float    v8f;

#define NB      4096
#define NSEQ    64
#define DMODEL  256
#define NHEADS  8
#define HEADD   32
#define DFFN    1024

// workspace layout (in halves)
#define WQ      0              // Q f16 [64][256] row-major (A operand)
#define WVE2    16384          // ve_w2  swizzled (K=128,N=256)
#define WIN     49152          // attn_in_w swizzled (256,768)
#define WOUT    245760         // attn_out_w swizzled (256,256)
#define WF1     311296         // ffn_w1 swizzled (256,1024)
#define WF2     573440         // ffn_w2 swizzled (1024,256)
#define WKW     835584         // k_w swizzled (256,256)

// dynamic LDS: 64KB Hf + 96KB R2 + 96KB R3 + 32KB vT + 1KB scratch
#define SMEM_BYTES (65536 + 98304 + 98304 + 32768 + 1024)

__device__ inline v16h cat8(v8h lo, v8h hi) {
  return __builtin_shufflevector(lo, hi, 0, 1, 2, 3, 4, 5, 6, 7,
                                 8, 9, 10, 11, 12, 13, 14, 15);
}

// ---------------- WMMA operand loaders (wave32, 16x16x32 f16) --------------
// A 16x32 (MxK), 16-contiguous-K per lane half:
//   lanes 0-15  row M=lane   : halves 0-15 = K 0-15
//   lanes 16-31 row M=lane-16: halves 0-15 = K 16-31
__device__ inline v16h load_a16(const _Float16* p0, int ld, int m0, int k0, int lane) {
  int row = m0 + (lane & 15);
  int kb  = k0 + ((lane & 16) ? 16 : 0);
  const _Float16* p = p0 + row * ld + kb;      // 16B-aligned (k0 mult of 8)
  return cat8(*(const v8h*)p, *(const v8h*)(p + 8));
}

__device__ inline v16h load_a32(const float* p0, int ld, int m0, int k0, int lane) {
  int row = m0 + (lane & 15);
  int kb  = k0 + ((lane & 16) ? 16 : 0);
  const float* p = p0 + row * ld + kb;         // 32B-aligned
  v8f lo = *(const v8f*)p, hi = *(const v8f*)(p + 8);
  v16h a;
#pragma unroll
  for (int i = 0; i < 8; ++i) { a[i] = (_Float16)lo[i]; a[i + 8] = (_Float16)hi[i]; }
  return a;
}

// B 32x16 (KxN), 16-contiguous-K per lane half:
//   lanes 0-15 col N=lane: halves = K 0..15 ; lanes 16-31: K 16..31
// Swizzled-weight fetch: tiles stored as [tile][lane][16 halves] contiguous.
__device__ inline v16h load_b_swz(const _Float16* base, int tileIdx, int lane) {
  const _Float16* p = base + ((size_t)tileIdx * 32 + lane) * 16;
  return cat8(*(const v8h*)p, *(const v8h*)(p + 8));
}

// Transposed source: B[k][n] = src[n][k], src row-major [n][k] -> contiguous k
__device__ inline v16h load_bt(const _Float16* p0, int ld, int n0, int k0, int lane) {
  int row = n0 + (lane & 15);
  int kb  = k0 + ((lane & 16) ? 16 : 0);       // 32B-aligned (k0 mult of 32)
  const _Float16* p = p0 + row * ld + kb;
  return cat8(*(const v8h*)p, *(const v8h*)(p + 8));
}

#define WMMA(c, a, b) \
  __builtin_amdgcn_wmma_f32_16x16x32_f16(false, (a), false, (b), (short)0, (c), false, false)

// A f16 (lda) x swizzled weight B; NT = N/16 tiles per K-row of tiles
__device__ inline v8f mm_AB(const _Float16* A, int lda, int m0, int ka0,
                            const _Float16* Bs, int NT, int nt, int kt0,
                            int K, int lane, v8f c) {
  for (int kc = 0; kc < K; kc += 32) {
    v16h a = load_a16(A, lda, m0, ka0 + kc, lane);
    v16h b = load_b_swz(Bs, (kt0 + (kc >> 5)) * NT + nt, lane);
    c = WMMA(c, a, b);
  }
  return c;
}

// A f16 x B from transposed source (row-major [n][k])
__device__ inline v8f mm_ABt(const _Float16* A, int lda, int m0, int ka0,
                             const _Float16* Bt, int ldb, int n0, int kb0,
                             int K, int lane, v8f c) {
  for (int kc = 0; kc < K; kc += 32) {
    v16h a = load_a16(A, lda, m0, ka0 + kc, lane);
    v16h b = load_bt(Bt, ldb, n0, kb0 + kc, lane);
    c = WMMA(c, a, b);
  }
  return c;
}

// A f32 x B from transposed source (attn @ v via vT)
__device__ inline v8f mm_A32Bt(const float* A, int lda, int m0,
                               const _Float16* Bt, int ldb, int n0, int kb0,
                               int K, int lane, v8f c) {
  for (int kc = 0; kc < K; kc += 32) {
    v16h a = load_a32(A, lda, m0, kc, lane);
    v16h b = load_bt(Bt, ldb, n0, kb0 + kc, lane);
    c = WMMA(c, a, b);
  }
  return c;
}

// ---------------------------- prep kernels ---------------------------------
// Convert f32 weight [K][N] into f16 WMMA-B swizzled layout:
//   dst[((kt*NT + nt)*32 + lane)*16 + e], lane = (kl>=16?16:0)|nl, e = kl&15
__global__ void cvt_swz_kernel(const float* __restrict__ src,
                               _Float16* __restrict__ dst, int K, int N) {
  int i = blockIdx.x * blockDim.x + threadIdx.x;
  if (i >= K * N) return;
  int k = i / N, n = i - k * N;
  int kt = k >> 5, kl = k & 31;
  int nt = n >> 4, nl = n & 15;
  int lane = ((kl >> 4) << 4) | nl;
  int e = kl & 15;
  int NT = N >> 4;
  dst[(((size_t)(kt * NT + nt) * 32) + lane) * 16 + e] = (_Float16)src[i];
}

// Q = rank_emb[:64] @ q_w + q_b -> f16 [64][256] row-major (runs once, tiny)
__global__ void qprep_kernel(const float* __restrict__ rank_emb,
                             const float* __restrict__ q_w,
                             const float* __restrict__ q_b,
                             _Float16* __restrict__ Qf) {
  int t = blockIdx.x, d = threadIdx.x;
  float acc = q_b[d];
  for (int j = 0; j < DMODEL; ++j) acc += rank_emb[t * DMODEL + j] * q_w[j * DMODEL + d];
  Qf[t * DMODEL + d] = (_Float16)acc;
}

// ----------------------------- main kernel ---------------------------------
__global__ __launch_bounds__(256)
void pointer_sort_kernel(const float* __restrict__ x,
                         const float* __restrict__ ve_w1,
                         const float* __restrict__ ve_b1,
                         const float* __restrict__ ve_b2,
                         const float* __restrict__ pos_emb,
                         const float* __restrict__ attn_in_b,
                         const float* __restrict__ attn_out_b,
                         const float* __restrict__ ffn_b1,
                         const float* __restrict__ ffn_b2,
                         const float* __restrict__ ln1_g, const float* __restrict__ ln1_b,
                         const float* __restrict__ ln2_g, const float* __restrict__ ln2_b,
                         const float* __restrict__ k_b,
                         const _Float16* __restrict__ ws,
                         float* __restrict__ out) {
  extern __shared__ char smem[];
  float* Hf = (float*)smem;                    // [64][256] fp32 residual stream
  char*  r2 = smem + 65536;                    // 96KB multiplexed
  char*  r3 = r2 + 98304;                      // 96KB multiplexed
  _Float16* vT = (_Float16*)(r3 + 98304);      // [8][32][64] v transposed, 32KB
  float* scratch = (float*)((char*)vT + 32768);

  _Float16* hid   = (_Float16*)r2;             // [64][128] MLP hidden
  float*    S     = (float*)r2;                // [4][64][64] scores (per pass)
  _Float16* hf16a = (_Float16*)(r2 + 65536);   // [64][256] h f16
  _Float16* aobuf = (_Float16*)(r2 + 65536);   // [64][256] attn out f16
  float*    Bs    = (float*)r2;                // [64][64] base logits
  _Float16* qkv   = (_Float16*)r3;             // [64][768]
  _Float16* hf16b = (_Float16*)r3;             // [64][256] post-LN activations
  _Float16* hid2  = (_Float16*)(r3 + 32768);   // [64][512] FFN hidden chunk
  _Float16* Kbuf  = (_Float16*)(r3 + 32768);   // [64][256] K f16

  const int tid  = threadIdx.x;
  const int lane = tid & 31;
  const int wave = tid >> 5;
  const int b    = blockIdx.x;

  // ---- phase 0a: hidden = relu(x * ve_w1 + ve_b1)  [64][128] ----
  for (int i = tid; i < NSEQ * 128; i += 256) {
    int row = i >> 7, j = i & 127;
    float hv = x[(size_t)b * NSEQ + row] * ve_w1[j] + ve_b1[j];
    hid[i] = (_Float16)fmaxf(hv, 0.f);
  }
  __syncthreads();

  // ---- phase 0b: h = hidden @ ve_w2 + ve_b2 + pos_emb -> Hf (f32), hf16a ----
  for (int tile = wave; tile < 64; tile += 8) {
    int m0 = (tile & 3) * 16, nt = tile >> 2, n0 = nt * 16;
    v8f c = {};
    c = mm_AB(hid, 128, m0, 0, ws + WVE2, 16, nt, 0, 128, lane, c);
    int col = n0 + (lane & 15);
    int rb  = m0 + ((lane & 16) ? 8 : 0);
    float bias = ve_b2[col];
#pragma unroll
    for (int r = 0; r < 8; ++r) {
      int row = rb + r;
      float v = c[r] + bias + pos_emb[row * DMODEL + col];
      Hf[row * DMODEL + col]    = v;
      hf16a[row * DMODEL + col] = (_Float16)v;
    }
  }
  __syncthreads();

  // ---- phase 1: qkv = h @ attn_in_w + b ; q pre-scaled; stage vT ----
  for (int tile = wave; tile < 192; tile += 8) {
    int m0 = (tile & 3) * 16, nt = tile >> 2, n0 = nt * 16;
    v8f c = {};
    c = mm_AB(hf16a, 256, m0, 0, ws + WIN, 48, nt, 0, 256, lane, c);
    int col = n0 + (lane & 15);
    int rb  = m0 + ((lane & 16) ? 8 : 0);
    float bias = attn_in_b[col];
    float scale = (col < DMODEL) ? 0.17677669529663687f : 1.0f;  // 1/sqrt(32)
    v8h pk;
#pragma unroll
    for (int r = 0; r < 8; ++r) {
      _Float16 hv = (_Float16)((c[r] + bias) * scale);
      qkv[(rb + r) * 768 + col] = hv;
      pk[r] = hv;
    }
    if (col >= 2 * DMODEL) {                    // v slice -> vT[head][dim][key]
      int dg = col - 2 * DMODEL;
      // keys rb..rb+7 contiguous -> single 16B store
      *(v8h*)(vT + (dg >> 5) * 2048 + (dg & 31) * 64 + rb) = pk;
    }
  }
  __syncthreads();

  // ---- phase 2: attention, two passes of 4 heads (2 waves per head) ----
  for (int pass = 0; pass < 2; ++pass) {
    int head = pass * 4 + (wave >> 1);
    int slot = wave >> 1;
    int rh   = wave & 1;                        // rows [rh*32, rh*32+32)
    // scores = q @ k^T  (K=32): B from k rows (transposed source)
    for (int mt = 0; mt < 2; ++mt) {
      int m0 = rh * 32 + mt * 16;
      for (int nt = 0; nt < 4; ++nt) {
        int n0 = nt * 16;
        v8f c = {};
        c = mm_ABt(qkv, 768, m0, head * HEADD,
                   qkv, 768, n0, DMODEL + head * HEADD, HEADD, lane, c);
        int col = n0 + (lane & 15);
        int rb  = m0 + ((lane & 16) ? 8 : 0);
#pragma unroll
        for (int r = 0; r < 8; ++r) S[slot * 4096 + (rb + r) * 64 + col] = c[r];
      }
    }
    __syncthreads();
    // softmax: 256 threads <-> 256 rows (4 slots x 64 rows)
    {
      float* p = S + (tid >> 6) * 4096 + (tid & 63) * 64;
      float m = p[0];
      for (int j = 1; j < 64; ++j) m = fmaxf(m, p[j]);
      float s = 0.f;
      for (int j = 0; j < 64; ++j) { float e = __expf(p[j] - m); p[j] = e; s += e; }
      float inv = 1.f / s;
      for (int j = 0; j < 64; ++j) p[j] *= inv;
    }
    __syncthreads();
    // ao = attn @ v  (K=64): B from vT[head] (transposed source, contiguous k)
    for (int mt = 0; mt < 2; ++mt) {
      int m0 = rh * 32 + mt * 16;
      for (int nt = 0; nt < 2; ++nt) {
        int n0 = nt * 16;
        v8f c = {};
        c = mm_A32Bt(S + slot * 4096, 64, m0,
                     vT + head * 2048, 64, n0, 0, 64, lane, c);
        int col = head * HEADD + n0 + (lane & 15);
        int rb  = m0 + ((lane & 16) ? 8 : 0);
#pragma unroll
        for (int r = 0; r < 8; ++r) aobuf[(rb + r) * DMODEL + col] = (_Float16)c[r];
      }
    }
    __syncthreads();
  }

  // ---- phase 3: Hf += ao @ attn_out_w + b ----
  for (int tile = wave; tile < 64; tile += 8) {
    int m0 = (tile & 3) * 16, nt = tile >> 2, n0 = nt * 16;
    v8f c = {};
    c = mm_AB(aobuf, 256, m0, 0, ws + WOUT, 16, nt, 0, 256, lane, c);
    int col = n0 + (lane & 15);
    int rb  = m0 + ((lane & 16) ? 8 : 0);
    float bias = attn_out_b[col];
#pragma unroll
    for (int r = 0; r < 8; ++r) Hf[(rb + r) * DMODEL + col] += c[r] + bias;
  }
  __syncthreads();

  // ---- LN1 -> Hf (h), hf16b ----
  if (tid < NSEQ) {
    float* row = Hf + tid * DMODEL;
    float m = 0.f;
    for (int j = 0; j < DMODEL; ++j) m += row[j];
    m *= (1.f / DMODEL);
    float v = 0.f;
    for (int j = 0; j < DMODEL; ++j) { float d = row[j] - m; v += d * d; }
    float inv = rsqrtf(v * (1.f / DMODEL) + 1e-5f);
    for (int j = 0; j < DMODEL; ++j) {
      float val = (row[j] - m) * inv * ln1_g[j] + ln1_b[j];
      row[j] = val;
      hf16b[tid * DMODEL + j] = (_Float16)val;
    }
  }
  __syncthreads();

  // ---- phase 4: FFN in two 512-wide chunks ----
  for (int c2 = 0; c2 < 2; ++c2) {
    // hid2 = relu(h @ ffn_w1[:, c2*512 : +512] + b1)
    for (int tile = wave; tile < 128; tile += 8) {
      int m0 = (tile & 3) * 16, ntl = tile >> 2, n0 = ntl * 16;
      v8f c = {};
      c = mm_AB(hf16b, 256, m0, 0, ws + WF1, 64, c2 * 32 + ntl, 0, 256, lane, c);
      int coll = n0 + (lane & 15);
      int colg = c2 * 512 + coll;
      int rb   = m0 + ((lane & 16) ? 8 : 0);
      float bias = ffn_b1[colg];
#pragma unroll
      for (int r = 0; r < 8; ++r)
        hid2[(rb + r) * 512 + coll] = (_Float16)fmaxf(c[r] + bias, 0.f);
    }
    __syncthreads();
    // Hf += hid2 @ ffn_w2[c2*512 : +512, :]  (+ b2 once)
    for (int tile = wave; tile < 64; tile += 8) {
      int m0 = (tile & 3) * 16, nt = tile >> 2, n0 = nt * 16;
      v8f c = {};
      c = mm_AB(hid2, 512, m0, 0, ws + WF2, 16, nt, c2 * 16, 512, lane, c);
      int col = n0 + (lane & 15);
      int rb  = m0 + ((lane & 16) ? 8 : 0);
      float bias = (c2 == 0) ? ffn_b2[col] : 0.f;
#pragma unroll
      for (int r = 0; r < 8; ++r) Hf[(rb + r) * DMODEL + col] += c[r] + bias;
    }
    __syncthreads();
  }

  // ---- LN2 -> Hf (enc), hf16b ----
  if (tid < NSEQ) {
    float* row = Hf + tid * DMODEL;
    float m = 0.f;
    for (int j = 0; j < DMODEL; ++j) m += row[j];
    m *= (1.f / DMODEL);
    float v = 0.f;
    for (int j = 0; j < DMODEL; ++j) { float d = row[j] - m; v += d * d; }
    float inv = rsqrtf(v * (1.f / DMODEL) + 1e-5f);
    for (int j = 0; j < DMODEL; ++j) {
      float val = (row[j] - m) * inv * ln2_g[j] + ln2_b[j];
      row[j] = val;
      hf16b[tid * DMODEL + j] = (_Float16)val;
    }
  }
  __syncthreads();

  // ---- phase 5: K = enc @ k_w + k_b -> Kbuf f16 ----
  for (int tile = wave; tile < 64; tile += 8) {
    int m0 = (tile & 3) * 16, nt = tile >> 2, n0 = nt * 16;
    v8f c = {};
    c = mm_AB(hf16b, 256, m0, 0, ws + WKW, 16, nt, 0, 256, lane, c);
    int col = n0 + (lane & 15);
    int rb  = m0 + ((lane & 16) ? 8 : 0);
    float bias = k_b[col];
#pragma unroll
    for (int r = 0; r < 8; ++r)
      Kbuf[(rb + r) * DMODEL + col] = (_Float16)(c[r] + bias);
  }
  __syncthreads();

  // ---- phase 6: base = Q @ K^T -> Bs [64(t)][64(n)] f32 ----
  for (int tile = wave; tile < 16; tile += 8) {
    int m0 = (tile & 3) * 16, n0 = (tile >> 2) * 16;
    v8f c = {};
    c = mm_ABt(ws + WQ, 256, m0, 0, Kbuf, 256, n0, 0, 256, lane, c);
    int col = n0 + (lane & 15);
    int rb  = m0 + ((lane & 16) ? 8 : 0);
#pragma unroll
    for (int r = 0; r < 8; ++r) Bs[(rb + r) * 64 + col] = c[r];
  }
  __syncthreads();

  // ---- phase 7: sequential greedy decode with coverage mask ----
  float* mk = scratch;        // mask[64]
  float* rv = scratch + 64;   // reduction values[64]
  if (tid < NSEQ) mk[tid] = 1.f;
  __syncthreads();
  const float NEGINF = -__builtin_inff();
  for (int t = 0; t < NSEQ; ++t) {
    if (tid < NSEQ) {
      float logit = (mk[tid] > 0.f) ? Bs[t * 64 + tid] : NEGINF;
      out[(((size_t)b * NSEQ) + t) * NSEQ + tid] = logit;
      rv[tid] = logit;
    }
    __syncthreads();
    if (tid == 0) {
      float best = rv[0];
      int arg = 0;
      for (int j = 1; j < NSEQ; ++j)
        if (rv[j] > best) { best = rv[j]; arg = j; }   // first max (jnp.argmax)
      mk[arg] = 0.f;
    }
    __syncthreads();
  }
}

// ------------------------------- launcher ----------------------------------
extern "C" void kernel_launch(void* const* d_in, const int* in_sizes, int n_in,
                              void* d_out, int out_size, void* d_ws, size_t ws_size,
                              hipStream_t stream) {
  _Float16* ws = (_Float16*)d_ws;
  // inputs (setup_inputs order):
  // 0 x, 1 ve_w1, 2 ve_b1, 3 ve_w2, 4 ve_b2, 5 pos_emb, 6 attn_in_w, 7 attn_in_b,
  // 8 attn_out_w, 9 attn_out_b, 10 ffn_w1, 11 ffn_b1, 12 ffn_w2, 13 ffn_b2,
  // 14 ln1_g, 15 ln1_b, 16 ln2_g, 17 ln2_b, 18 rank_emb, 19 q_w, 20 q_b, 21 k_w, 22 k_b
  auto swz = [&](int idx, size_t off, int K, int N) {
    int n = K * N;
    cvt_swz_kernel<<<(n + 255) / 256, 256, 0, stream>>>((const float*)d_in[idx],
                                                        ws + off, K, N);
  };
  swz(3,  WVE2, 128, 256);
  swz(6,  WIN,  256, 768);
  swz(8,  WOUT, 256, 256);
  swz(10, WF1,  256, 1024);
  swz(12, WF2,  1024, 256);
  swz(21, WKW,  256, 256);
  qprep_kernel<<<NSEQ, DMODEL, 0, stream>>>((const float*)d_in[18],
                                            (const float*)d_in[19],
                                            (const float*)d_in[20], ws + WQ);

  (void)hipFuncSetAttribute((const void*)pointer_sort_kernel,
                            hipFuncAttributeMaxDynamicSharedMemorySize, SMEM_BYTES);

  pointer_sort_kernel<<<NB, 256, SMEM_BYTES, stream>>>(
      (const float*)d_in[0],  (const float*)d_in[1],  (const float*)d_in[2],
      (const float*)d_in[4],  (const float*)d_in[5],  (const float*)d_in[7],
      (const float*)d_in[9],  (const float*)d_in[11], (const float*)d_in[13],
      (const float*)d_in[14], (const float*)d_in[15], (const float*)d_in[16],
      (const float*)d_in[17], (const float*)d_in[22], ws, (float*)d_out);
}
